// MultiHeadedAttention_65111704207486
// MI455X (gfx1250) — matrix operand order repaired and checked by reading, not compiled, not run on hardware
//
#include <hip/hip_runtime.h>

// ---------------------------------------------------------------------------
// MHA for MI455X (gfx1250, wave32). All matmuls via v_wmma_f32_16x16x32_bf16.
// Flash attention computes S^T = K*Q^T and O^T = V^T*P^T so the softmax
// reduction is in-lane (k on the VGPR/m axis, q on the lane/n axis): only one
// cross-half shuffle per reduction, per-lane scalar running max/sum.
// ---------------------------------------------------------------------------

#define DEV __device__ __forceinline__

typedef __attribute__((ext_vector_type(8)))  float  v8f;
typedef __attribute__((ext_vector_type(8)))  __bf16 v8bf;
typedef __attribute__((ext_vector_type(16))) __bf16 v16bf;

static constexpr int Bsz = 2, Hh = 8, Ss = 4096, Dk = 64, Dm = 512;
static constexpr int Mrows = Bsz * Ss; // 8192 token rows

DEV __bf16 f2bf(float f) {
  union { float f; unsigned u; } v; v.f = f;
  unsigned r = v.u + 0x7FFFu + ((v.u >> 16) & 1u); // round-to-nearest-even
  unsigned short h = (unsigned short)(r >> 16);
  return __builtin_bit_cast(__bf16, h);
}

DEV v8f wmma_bf16(v16bf a, v16bf b, v8f c) {
  // (neg_a, A, neg_b, B, c_mod, C, reuse_a, reuse_b)
  return __builtin_amdgcn_wmma_f32_16x16x32_bf16(false, a, false, b, (short)0, c,
                                                 false, false);
}

DEV v16bf cat8(v8bf lo, v8bf hi) {
  v16bf r;
#pragma unroll
  for (int i = 0; i < 8; ++i) { r[i] = lo[i]; r[i + 8] = hi[i]; }
  return r;
}

// A-fragment (16x32 bf16, M=lane%16): 16 contiguous elems at K=lhalf*16
// of a row-major [M][32] tile.
DEV v16bf load_afrag(const __bf16* rowbase, int lhalf) {
  const __bf16* p = rowbase + lhalf * 16;
  return cat8(*(const v8bf*)p, *(const v8bf*)(p + 8));
}

// ---------------- conversion kernels ----------------
__global__ void k_cvt_bf16(const float* __restrict__ s, __bf16* __restrict__ d, int n) {
  int i = blockIdx.x * blockDim.x + threadIdx.x;
  if (i < n) d[i] = f2bf(s[i]);
}

// W stored [in][out] fp32 -> WT [out][in] bf16 so B-fragments are contiguous.
__global__ void k_transpose_w(const float* __restrict__ W, __bf16* __restrict__ WT) {
  int i = blockIdx.x * blockDim.x + threadIdx.x; // Dm*Dm threads
  int o  = i & (Dm - 1);
  int in = i >> 9;
  WT[(size_t)o * Dm + in] = f2bf(W[i]); // i = in*Dm + o
}

// ---------------- bf16 WMMA GEMM: C[16x64 per wave] = A[M,512] * WT^T + bias ---
// MODE 0: -> Qh [b,h,s,d] bf16    MODE 1: -> Kh [b,h,s,d] bf16
// MODE 2: -> VT [b,h,d,s] bf16    MODE 3: -> out fp32 [m, n]
template <int MODE>
__global__ void k_gemm(const __bf16* __restrict__ A, const __bf16* __restrict__ WT,
                       const float* __restrict__ bias, void* __restrict__ out) {
  const int lane = threadIdx.x & 31, wave = threadIdx.x >> 5;
  const int lrow = lane & 15, lhalf = lane >> 4;
  const int gt = blockIdx.x * 8 + wave;          // 4096 tiles
  const int m0 = (gt >> 3) << 4;                 // 512 M-tiles of 16
  const int n0 = (gt & 7) << 6;                  // 8 N-tiles of 64

  v8f acc[4];
#pragma unroll
  for (int t = 0; t < 4; ++t) {
    float bv = bias[n0 + t * 16 + lrow];         // bias is constant along M
#pragma unroll
    for (int r = 0; r < 8; ++r) acc[t][r] = bv;
  }

  const __bf16* arow = A + (size_t)(m0 + lrow) * Dm;
#pragma unroll 4
  for (int k0 = 0; k0 < Dm; k0 += 32) {
    v16bf af = load_afrag(arow + k0, lhalf);
#pragma unroll
    for (int t = 0; t < 4; ++t) {
      v16bf bf = *(const v16bf*)(WT + (size_t)(n0 + t * 16 + lrow) * Dm +
                                 k0 + lhalf * 16);
      acc[t] = wmma_bf16(af, bf, acc[t]);
    }
  }

#pragma unroll
  for (int r = 0; r < 8; ++r) {
    int m = m0 + r + 8 * lhalf;                  // C layout: vgpr=m%8, half=m/8
    int b = m >> 12, s = m & (Ss - 1);
#pragma unroll
    for (int t = 0; t < 4; ++t) {
      int n = n0 + t * 16 + lrow;
      float vl = acc[t][r];
      if (MODE == 3) {
        ((float*)out)[(size_t)m * Dm + n] = vl;
      } else {
        int h = n >> 6, d = n & 63;
        __bf16* o = (__bf16*)out;
        if (MODE == 2)
          o[(((size_t)(b * Hh + h)) * Dk + d) * Ss + s] = f2bf(vl); // V^T
        else
          o[(((size_t)(b * Hh + h)) * Ss + s) * Dk + d] = f2bf(vl); // Q / K
      }
    }
  }
}

// ---------------- flash attention (transposed): one wave = 16 q, dk=64 -------
__global__ void k_flash(const __bf16* __restrict__ Qh, const __bf16* __restrict__ Kh,
                        const __bf16* __restrict__ VTh, __bf16* __restrict__ ctxb) {
  __shared__ __bf16 plds[8][16 * 32];            // per-wave P tile [q][k], 8 KB
  const int lane = threadIdx.x & 31, wave = threadIdx.x >> 5;
  const int lrow = lane & 15, lhalf = lane >> 4;
  const int gt = blockIdx.x * 8 + wave;          // 4096 q-tiles
  const int bh = gt >> 8;                        // (b*H + h)
  const int q0 = (gt & 255) << 4;

  const __bf16* Qp  = Qh  + ((size_t)bh * Ss + q0) * Dk;
  const __bf16* Kp  = Kh  + (size_t)bh * Ss * Dk;
  const __bf16* Vtp = VTh + (size_t)bh * Dk * Ss;
  __bf16* pl = plds[wave];

  // Q^T as two resident B-fragments: lane holds q=lrow, d = c*32+lhalf*16..+15
  v16bf qb[2];
#pragma unroll
  for (int c = 0; c < 2; ++c)
    qb[c] = *(const v16bf*)(Qp + lrow * Dk + c * 32 + lhalf * 16);

  v8f acc[4];                                    // O^T: d on m-axis, q on lanes
#pragma unroll
  for (int t = 0; t < 4; ++t)
#pragma unroll
    for (int r = 0; r < 8; ++r) acc[t][r] = 0.f;
  float mrun = -1e30f, lrun = 0.f;               // per-lane (per-q) scalars!

  for (int sk0 = 0; sk0 < Ss; sk0 += 32) {
    // ---- S^T(32k x 16q) = K * Q^T : 4 WMMAs, k on m-axis ----
    v8f sc[2];
#pragma unroll
    for (int kt = 0; kt < 2; ++kt) {             // two 16-row K tiles
      const __bf16* krow = Kp + (size_t)(sk0 + kt * 16 + lrow) * Dk;
      v8f z;
#pragma unroll
      for (int r = 0; r < 8; ++r) z[r] = 0.f;
      z      = wmma_bf16(load_afrag(krow,      lhalf), qb[0], z);
      sc[kt] = wmma_bf16(load_afrag(krow + 32, lhalf), qb[1], z);
    }
    // ---- online softmax: column of S^T == softmax row; mostly in-lane ----
    float mloc = -1e30f;
#pragma unroll
    for (int kt = 0; kt < 2; ++kt)
#pragma unroll
      for (int r = 0; r < 8; ++r) {
        float s = sc[kt][r] * 0.125f;            // 1/sqrt(dk)
        sc[kt][r] = s;
        mloc = fmaxf(mloc, s);
      }
    mloc = fmaxf(mloc, __shfl_xor(mloc, 16, 32)); // single cross-half exchange
    float mnew  = fmaxf(mrun, mloc);
    float alpha = __expf(mrun - mnew);
    float ssum = 0.f;
#pragma unroll
    for (int kt = 0; kt < 2; ++kt) {
      v8bf pk;
#pragma unroll
      for (int r = 0; r < 8; ++r) {
        float p = __expf(sc[kt][r] - mnew);
        ssum += p;
        pk[r] = f2bf(p);
      }
      // P[q][k]: lane writes 8 consecutive k = kt*16 + lhalf*8 + r
      *(v8bf*)(pl + lrow * 32 + kt * 16 + lhalf * 8) = pk;
    }
    ssum += __shfl_xor(ssum, 16, 32);
    lrun = lrun * alpha + ssum;
    mrun = mnew;
#pragma unroll
    for (int t = 0; t < 4; ++t)
#pragma unroll
      for (int r = 0; r < 8; ++r) acc[t][r] *= alpha;

    // ---- O^T += V^T * P^T : P^T as B-frag straight from LDS [q][k] ----
    v16bf pb = *(const v16bf*)(pl + lrow * 32 + lhalf * 16); // contiguous 32B
#pragma unroll
    for (int t = 0; t < 4; ++t) {
      const __bf16* vrow = Vtp + (size_t)(t * 16 + lrow) * Ss + sk0;
      acc[t] = wmma_bf16(load_afrag(vrow, lhalf), pb, acc[t]);
    }
  }

  // ---- epilogue: lane = q, vgpr axis = d -> contiguous 16B stores ----
  const int b = bh >> 3, h = bh & 7;
  const float inv = 1.f / lrun;
  __bf16* orow = ctxb + ((size_t)b * Ss + (q0 + lrow)) * Dm + (size_t)h * Dk;
#pragma unroll
  for (int t = 0; t < 4; ++t) {
    v8bf ov;
#pragma unroll
    for (int r = 0; r < 8; ++r) ov[r] = f2bf(acc[t][r] * inv);
    *(v8bf*)(orow + t * 16 + 8 * lhalf) = ov;    // d = t*16 + 8*lhalf + r
  }
}

// ---------------------------------------------------------------------------
extern "C" void kernel_launch(void* const* d_in, const int* in_sizes, int n_in,
                              void* d_out, int out_size, void* d_ws, size_t ws_size,
                              hipStream_t stream) {
  const float* q  = (const float*)d_in[0];
  const float* k  = (const float*)d_in[1];
  const float* v  = (const float*)d_in[2];
  const float* Wq = (const float*)d_in[3];
  const float* bq = (const float*)d_in[4];
  const float* Wk = (const float*)d_in[5];
  const float* bk = (const float*)d_in[6];
  const float* Wv = (const float*)d_in[7];
  const float* bv = (const float*)d_in[8];
  const float* Wo = (const float*)d_in[9];
  const float* bo = (const float*)d_in[10];

  char* ws = (char*)d_ws;
  size_t off = 0;
  auto alloc = [&](size_t bytes) -> void* {
    void* p = ws + off;
    off = (off + bytes + 255) & ~(size_t)255;
    return p;
  };
  const size_t actN = (size_t)Mrows * Dm;  // 8192*512 elements
  const size_t wN   = (size_t)Dm * Dm;     // 512*512
  __bf16* xq  = (__bf16*)alloc(actN * 2);
  __bf16* xk  = (__bf16*)alloc(actN * 2);
  __bf16* xv  = (__bf16*)alloc(actN * 2);
  __bf16* WqT = (__bf16*)alloc(wN * 2);
  __bf16* WkT = (__bf16*)alloc(wN * 2);
  __bf16* WvT = (__bf16*)alloc(wN * 2);
  __bf16* WoT = (__bf16*)alloc(wN * 2);
  __bf16* Qh  = (__bf16*)alloc(actN * 2);
  __bf16* Kh  = (__bf16*)alloc(actN * 2);
  __bf16* VTh = (__bf16*)alloc(actN * 2);
  __bf16* ctb = (__bf16*)alloc(actN * 2);

  // 1) precision conversion + weight transpose
  int cvtBlocks = (int)(actN / 256);
  k_cvt_bf16<<<cvtBlocks, 256, 0, stream>>>(q, xq, (int)actN);
  k_cvt_bf16<<<cvtBlocks, 256, 0, stream>>>(k, xk, (int)actN);
  k_cvt_bf16<<<cvtBlocks, 256, 0, stream>>>(v, xv, (int)actN);
  int wBlocks = (int)(wN / 256);
  k_transpose_w<<<wBlocks, 256, 0, stream>>>(Wq, WqT);
  k_transpose_w<<<wBlocks, 256, 0, stream>>>(Wk, WkT);
  k_transpose_w<<<wBlocks, 256, 0, stream>>>(Wv, WvT);
  k_transpose_w<<<wBlocks, 256, 0, stream>>>(Wo, WoT);

  // 2) QKV projections (WMMA GEMMs), head-split layouts
  k_gemm<0><<<512, 256, 0, stream>>>(xq, WqT, bq, Qh);
  k_gemm<1><<<512, 256, 0, stream>>>(xk, WkT, bk, Kh);
  k_gemm<2><<<512, 256, 0, stream>>>(xv, WvT, bv, VTh);

  // 3) flash attention (transposed layout)
  k_flash<<<512, 256, 0, stream>>>(Qh, Kh, VTh, ctb);

  // 4) output projection, fp32 result
  k_gemm<3><<<512, 256, 0, stream>>>(ctb, WoT, bo, d_out);
}
